// TrajectoryTransformer_17154099380295
// MI455X (gfx1250) — matrix operand order repaired and checked by reading, not compiled, not run on hardware
//
#include <hip/hip_runtime.h>
#include <math.h>

// ---------------------------------------------------------------------------
// Model constants (fixed by the reference)
// ---------------------------------------------------------------------------
#define Dm    256
#define Hh    8
#define DH    32
#define Lnum  2
#define Ff    2048
#define Bb    32
#define TMAX  200

#define NWAVES 16
#define BLOCK  (NWAVES * 32)   // 512 threads -> 4 waves/SIMD, big VGPR budget
#define HS 264                 // f32 LDS row stride (floats)
#define FS 2064                // f16 LDS row stride (halves) for FFN intermediate

typedef __attribute__((ext_vector_type(16))) _Float16 v16h;
typedef __attribute__((ext_vector_type(8)))  float    v8f;

// packed-weight offsets (in halves)
constexpr size_t SZ_QKV = (size_t)Dm * (3 * Dm);
constexpr size_t SZ_WO  = (size_t)Dm * Dm;
constexpr size_t SZ_W1  = (size_t)Dm * Ff;
constexpr size_t SZ_W2  = (size_t)Ff * Dm;
constexpr size_t SZ_CKV = (size_t)Dm * (2 * Dm);

constexpr size_t OFF_ENC_WQKV = 0;
constexpr size_t OFF_ENC_WO   = OFF_ENC_WQKV + 2 * SZ_QKV;
constexpr size_t OFF_ENC_W1   = OFF_ENC_WO   + 2 * SZ_WO;
constexpr size_t OFF_ENC_W2   = OFF_ENC_W1   + 2 * SZ_W1;
constexpr size_t OFF_DEC_WQKV = OFF_ENC_W2   + 2 * SZ_W2;
constexpr size_t OFF_DEC_WO   = OFF_DEC_WQKV + 2 * SZ_QKV;
constexpr size_t OFF_DEC_CWKV = OFF_DEC_WO   + 2 * SZ_WO;
constexpr size_t OFF_DEC_CWO  = OFF_DEC_CWKV + 2 * SZ_CKV;
constexpr size_t OFF_DEC_W1   = OFF_DEC_CWO  + 2 * SZ_WO;
constexpr size_t OFF_DEC_W2   = OFF_DEC_W1   + 2 * SZ_W1;
constexpr size_t W_TOTAL      = OFF_DEC_W2   + 2 * SZ_W2;

constexpr size_t LDS_BYTES = (size_t)3 * 32 * HS * 4 + (size_t)32 * FS * 2; // 233472

struct NetArgs {
  const float *ja, *jp, *tp;
  const int   *maxlen;
  const float *W_in, *b_in, *W_act, *b_act, *W_out, *b_out, *start_token;
  const float *enc_bqkv, *enc_bo, *enc_b1, *enc_b2;
  const float *enc_ln1_g, *enc_ln1_b, *enc_ln2_g, *enc_ln2_b;
  const float *dec_bqkv, *dec_bo, *dec_cbkv, *dec_cbo, *dec_b1, *dec_b2;
  const float *dec_ln1_g, *dec_ln1_b, *dec_ln2_g, *dec_ln2_b, *dec_ln3_g, *dec_ln3_b;
  const _Float16 *wpack;
  const float *pe;
  float *cross, *kc, *vc;
  float *out;
};

// ---------------------------------------------------------------------------
// Prep kernel 1: sinusoidal positional encoding table [TMAX, 256]
// ---------------------------------------------------------------------------
__global__ void pe_kernel(float* pe) {
  for (int idx = blockIdx.x * blockDim.x + threadIdx.x; idx < TMAX * Dm;
       idx += gridDim.x * blockDim.x) {
    int t = idx >> 8, d = idx & 255;
    int i = d >> 1;
    float div = expf(-0.07195578439406394f * (float)i);  // exp(-2i*ln(1e4)/256)
    float ang = (float)t * div;
    pe[idx] = (d & 1) ? cosf(ang) : sinf(ang);
  }
}

// ---------------------------------------------------------------------------
// Prep kernel 2: pack a K x N f32 weight matrix into f16 WMMA B-fragments.
// Fragment layout (halves): idx = (((nt*KT + kt)*32 + lane)*16 + e)
//   n = nt*16 + (lane & 15)
//   k = kt*32 + (e>=8 ? 16 : 0) + (lane>=16 ? 8 : 0) + (e & 7)   (ISA 16-bit B 32x16 layout)
// ---------------------------------------------------------------------------
__global__ void pack_b(const float* __restrict__ src, _Float16* __restrict__ dst,
                       int K, int N) {
  int KT = K >> 5;
  size_t total = (size_t)K * N;
  for (size_t idx = (size_t)blockIdx.x * blockDim.x + threadIdx.x; idx < total;
       idx += (size_t)gridDim.x * blockDim.x) {
    int e    = (int)(idx & 15);
    int lane = (int)((idx >> 4) & 31);
    size_t rest = idx >> 9;
    int kt = (int)(rest % KT);
    int nt = (int)(rest / KT);
    int k = kt * 32 + ((e >> 3) << 4) + ((lane >> 4) << 3) + (e & 7);
    int n = nt * 16 + (lane & 15);
    dst[idx] = (_Float16)src[(size_t)k * N + n];
  }
}

// ---------------------------------------------------------------------------
// Device helpers
// ---------------------------------------------------------------------------
__device__ inline float waveRed(float v) {
#pragma unroll
  for (int off = 16; off > 0; off >>= 1) v += __shfl_xor(v, off, 32);
  return v;
}

// WGP-scope speculative prefetch (default scope of global_prefetch_b8 is WGP:
// pulls into all cache levels; translation failures are silently dropped).
__device__ inline void prefetch_wgp(const void* p) {
  asm volatile("global_prefetch_b8 %0, off" :: "v"(p));
}

// A-fragment (16x32 f16) gathered from an f32 LDS buffer (row stride HS).
__device__ inline v16h load_a_f32(const float* src, int m0, int k0, int lane) {
  const float* p = src + (size_t)(m0 + (lane & 15)) * HS + k0 + ((lane >> 4) << 3);
  v16h a;
#pragma unroll
  for (int e = 0; e < 8; ++e) {
    a[e]     = (_Float16)p[e];
    a[e + 8] = (_Float16)p[e + 16];
  }
  return a;
}

// Same but from an f16 LDS buffer (row stride FS).
__device__ inline v16h load_a_f16(const _Float16* src, int m0, int k0, int lane) {
  const _Float16* p = src + (size_t)(m0 + (lane & 15)) * FS + k0 + ((lane >> 4) << 3);
  v16h a;
#pragma unroll
  for (int e = 0; e < 8; ++e) {
    a[e]     = p[e];
    a[e + 8] = p[e + 16];
  }
  return a;
}

// M=32 GEMM with register-cached A fragments (K=32*KT, KT<=8).
// Wave w owns mt = w&1 permanently; n-tiles strided by 8 across wave pairs,
// so the KT A-fragments are loaded ONCE per wave and reused for every n-tile.
// B fragments for a tile are batch-preloaded into registers (one load clause,
// then KT back-to-back WMMAs); next tile's B is prefetched at WGP scope.
template <int KT, typename ALoad, typename Epi>
__device__ inline void gemmA(ALoad aload, const _Float16* __restrict__ wfrag,
                             int nt0, int ntCount, const float* __restrict__ bias,
                             Epi epi) {
  const int wave = threadIdx.x >> 5;
  const int lane = threadIdx.x & 31;
  const int mt = wave & 1;
  v16h afr[KT];
#pragma unroll
  for (int kt = 0; kt < KT; ++kt) afr[kt] = aload(mt << 4, kt << 5, lane);
  for (int i = wave >> 1; i < ntCount; i += NWAVES / 2) {
    const int nt = nt0 + i;
    const _Float16* bp = wfrag + ((size_t)nt * KT * 32 + lane) * 16;
    // unconditional WGP-scope prefetch of this wave's NEXT tile
    const _Float16* np = bp + (size_t)(NWAVES / 2) * KT * 512;
#pragma unroll
    for (int kt = 0; kt < KT; ++kt) prefetch_wgp(np + (size_t)kt * 512);
    v16h bfr[KT];
#pragma unroll
    for (int kt = 0; kt < KT; ++kt)
      bfr[kt] = *(const v16h*)(bp + (size_t)kt * 512);
    const float bv = bias[nt * 16 + (lane & 15)];
    v8f c = {bv, bv, bv, bv, bv, bv, bv, bv};
#pragma unroll
    for (int kt = 0; kt < KT; ++kt)
      c = __builtin_amdgcn_wmma_f32_16x16x32_f16(false, afr[kt], false, bfr[kt],
                                                 (short)0, c, false, false);
    epi(c, mt, nt, lane);
  }
}

// Specialized K=2048 GEMM (W2): ntCount=16 -> exactly 2 n-tiles per wave.
// A fragments (from f16 LDS) are loaded in chunks of 8 and reused for BOTH
// tiles; B fragments batch-preloaded per chunk; two independent WMMA chains.
template <typename Epi>
__device__ inline void gemmW2(const _Float16* __restrict__ ffnbuf,
                              const _Float16* __restrict__ wfrag,
                              const float* __restrict__ bias, Epi epi) {
  const int wave = threadIdx.x >> 5;
  const int lane = threadIdx.x & 31;
  const int mt = wave & 1;
  const int i0 = wave >> 1;   // n-tile 0
  const int i1 = i0 + 8;      // n-tile 1
  const float bv0 = bias[i0 * 16 + (lane & 15)];
  const float bv1 = bias[i1 * 16 + (lane & 15)];
  v8f c0 = {bv0, bv0, bv0, bv0, bv0, bv0, bv0, bv0};
  v8f c1 = {bv1, bv1, bv1, bv1, bv1, bv1, bv1, bv1};
  const _Float16* bp0 = wfrag + ((size_t)i0 * 64 * 32 + lane) * 16;
  const _Float16* bp1 = wfrag + ((size_t)i1 * 64 * 32 + lane) * 16;
  for (int kc = 0; kc < 8; ++kc) {
    // prefetch next chunk of both B streams (harmless past-the-end on last)
#pragma unroll
    for (int k = 0; k < 8; ++k) {
      prefetch_wgp(bp0 + (size_t)(kc * 8 + 8 + k) * 512);
      prefetch_wgp(bp1 + (size_t)(kc * 8 + 8 + k) * 512);
    }
    v16h afr[8];
#pragma unroll
    for (int k = 0; k < 8; ++k)
      afr[k] = load_a_f16(ffnbuf, mt << 4, (kc * 8 + k) << 5, lane);
    v16h bfr[8];
#pragma unroll
    for (int k = 0; k < 8; ++k)
      bfr[k] = *(const v16h*)(bp0 + (size_t)(kc * 8 + k) * 512);
#pragma unroll
    for (int k = 0; k < 8; ++k)
      c0 = __builtin_amdgcn_wmma_f32_16x16x32_f16(false, afr[k], false, bfr[k],
                                                  (short)0, c0, false, false);
#pragma unroll
    for (int k = 0; k < 8; ++k)
      bfr[k] = *(const v16h*)(bp1 + (size_t)(kc * 8 + k) * 512);
#pragma unroll
    for (int k = 0; k < 8; ++k)
      c1 = __builtin_amdgcn_wmma_f32_16x16x32_f16(false, afr[k], false, bfr[k],
                                                  (short)0, c1, false, false);
  }
  epi(c0, mt, i0, lane);
  epi(c1, mt, i1, lane);
}

// LayerNorm over 256 columns of 32 rows; each of the 16 waves does 2 rows.
// Optional fused residual add from a global row-major [32][256] source.
__device__ inline void layernorm_add(float* buf, const float* __restrict__ g,
                                     const float* __restrict__ bb,
                                     const float* __restrict__ addsrc) {
  __syncthreads();
  const int wv = threadIdx.x >> 5, lane = threadIdx.x & 31;
#pragma unroll
  for (int rr = 0; rr < 2; ++rr) {
    const int w = wv + rr * 16;
    float* row = buf + (size_t)w * HS;
    if (addsrc) {
#pragma unroll
      for (int i = 0; i < 8; ++i) {
        int d = lane + i * 32;
        row[d] += addsrc[(size_t)w * Dm + d];
      }
    }
    float s = 0.f;
#pragma unroll
    for (int i = 0; i < 8; ++i) s += row[lane + i * 32];
    const float m = waveRed(s) * (1.f / 256.f);
    float v = 0.f;
#pragma unroll
    for (int i = 0; i < 8; ++i) {
      float x = row[lane + i * 32] - m;
      v += x * x;
    }
    const float var = waveRed(v) * (1.f / 256.f);
    const float r = rsqrtf(var + 1e-5f);
#pragma unroll
    for (int i = 0; i < 8; ++i) {
      int d = lane + i * 32;
      row[d] = (row[d] - m) * r * g[d] + bb[d];
    }
  }
  __syncthreads();
}
__device__ inline void layernorm(float* buf, const float* g, const float* bb) {
  layernorm_add(buf, g, bb, nullptr);
}

// ---------------------------------------------------------------------------
// Persistent kernel: encoder + cross precompute + 200-step KV-cached decode.
// One workgroup, 16 wave32 waves, one WGP.
// ---------------------------------------------------------------------------
__global__ void __launch_bounds__(BLOCK, 1) traj_main(NetArgs A) {
  extern __shared__ __align__(16) char smem[];
  float*    h   = (float*)smem;                 // [32][HS] activations
  float*    tok = h   + 32 * HS;                // [32][HS] decoder token
  float*    tmp = tok + 32 * HS;                // [32][HS] scratch
  _Float16* ffn = (_Float16*)(tmp + 32 * HS);   // [32][FS] FFN intermediate (f16)
  float*    ascr = (float*)ffn;                 // attention merge scratch (aliases ffn)

  const int tid = threadIdx.x;

  // ---- encoder input: x = concat(ja, jp, tp) @ W_in + b_in  (K=31, scalar) ----
  for (int i = tid; i < Bb * Dm; i += BLOCK) {
    int b = i >> 8, d = i & 255;
    float s = A.b_in[d];
#pragma unroll
    for (int k = 0; k < 7; ++k)  s += A.ja[b * 7 + k]  * A.W_in[k * Dm + d];
#pragma unroll
    for (int k = 0; k < 21; ++k) s += A.jp[b * 21 + k] * A.W_in[(7 + k) * Dm + d];
#pragma unroll
    for (int k = 0; k < 3; ++k)  s += A.tp[b * 3 + k]  * A.W_in[(28 + k) * Dm + d];
    h[(size_t)b * HS + d] = s;
  }
  __syncthreads();

  auto aH = [&](int m0, int k0, int ln) -> v16h { return load_a_f32(h,   m0, k0, ln); };
  auto aT = [&](int m0, int k0, int ln) -> v16h { return load_a_f32(tmp, m0, k0, ln); };

  // ---- encoder layers (seq_len==1: self-attn == Wo(V(x))) ----
  for (int l = 0; l < Lnum; ++l) {
    gemmA<8>(aH, A.wpack + OFF_ENC_WQKV + l * SZ_QKV, 32, 16,
             A.enc_bqkv + l * 3 * Dm,
             [&](v8f c, int mt, int nt, int ln) {
               int coln = (nt - 32) * 16 + (ln & 15);
               int mb = mt * 16 + ((ln >> 4) << 3);
#pragma unroll
               for (int r = 0; r < 8; ++r) tmp[(size_t)(mb + r) * HS + coln] = c[r];
             });
    __syncthreads();
    gemmA<8>(aT, A.wpack + OFF_ENC_WO + l * SZ_WO, 0, 16, A.enc_bo + l * Dm,
             [&](v8f c, int mt, int nt, int ln) {
               int col = nt * 16 + (ln & 15);
               int mb = mt * 16 + ((ln >> 4) << 3);
#pragma unroll
               for (int r = 0; r < 8; ++r) h[(size_t)(mb + r) * HS + col] += c[r];
             });
    layernorm(h, A.enc_ln1_g + l * Dm, A.enc_ln1_b + l * Dm);
    gemmA<8>(aH, A.wpack + OFF_ENC_W1 + l * SZ_W1, 0, 128, A.enc_b1 + l * Ff,
             [&](v8f c, int mt, int nt, int ln) {
               int col = nt * 16 + (ln & 15);
               int mb = mt * 16 + ((ln >> 4) << 3);
#pragma unroll
               for (int r = 0; r < 8; ++r)
                 ffn[(size_t)(mb + r) * FS + col] = (_Float16)fmaxf(c[r], 0.f);
             });
    __syncthreads();
    gemmW2(ffn, A.wpack + OFF_ENC_W2 + l * SZ_W2, A.enc_b2 + l * Dm,
           [&](v8f c, int mt, int nt, int ln) {
             int col = nt * 16 + (ln & 15);
             int mb = mt * 16 + ((ln >> 4) << 3);
#pragma unroll
             for (int r = 0; r < 8; ++r) h[(size_t)(mb + r) * HS + col] += c[r];
           });
    layernorm(h, A.enc_ln2_g + l * Dm, A.enc_ln2_b + l * Dm);
  }

  // ---- cross precompute: cross[l] = Wo_c(V_c(memory)) ----
  for (int l = 0; l < Lnum; ++l) {
    gemmA<8>(aH, A.wpack + OFF_DEC_CWKV + l * SZ_CKV, 16, 16,
             A.dec_cbkv + l * 2 * Dm,
             [&](v8f c, int mt, int nt, int ln) {
               int coln = (nt - 16) * 16 + (ln & 15);
               int mb = mt * 16 + ((ln >> 4) << 3);
#pragma unroll
               for (int r = 0; r < 8; ++r) tmp[(size_t)(mb + r) * HS + coln] = c[r];
             });
    __syncthreads();
    float* crossL = A.cross + (size_t)l * Bb * Dm;
    gemmA<8>(aT, A.wpack + OFF_DEC_CWO + l * SZ_WO, 0, 16, A.dec_cbo + l * Dm,
             [&](v8f c, int mt, int nt, int ln) {
               int col = nt * 16 + (ln & 15);
               int mb = mt * 16 + ((ln >> 4) << 3);
#pragma unroll
               for (int r = 0; r < 8; ++r) crossL[(size_t)(mb + r) * Dm + col] = c[r];
             });
    __threadfence();
    __syncthreads();
  }

  // ---- decode loop ----
  for (int i = tid; i < Bb * Dm; i += BLOCK)
    tok[(size_t)(i >> 8) * HS + (i & 255)] = A.start_token[i & 255];
  __syncthreads();

  int T = A.maxlen[0];
  if (T > TMAX) T = TMAX;
  if (T < 1) T = 1;

  for (int t = 0; t < T; ++t) {
    for (int i = tid; i < Bb * Dm; i += BLOCK) {
      int b = i >> 8, d = i & 255;
      h[(size_t)b * HS + d] = tok[(size_t)b * HS + d] + A.pe[t * Dm + d];
    }
    __syncthreads();

    for (int l = 0; l < Lnum; ++l) {
      float* kcL = A.kc + (size_t)l * Bb * Hh * TMAX * DH;
      float* vcL = A.vc + (size_t)l * Bb * Hh * TMAX * DH;

      // qkv = h @ Wqkv + b : q -> tmp LDS, k/v -> caches (global, L2-resident)
      gemmA<8>(aH, A.wpack + OFF_DEC_WQKV + l * SZ_QKV, 0, 48,
               A.dec_bqkv + l * 3 * Dm,
               [&](v8f c, int mt, int nt, int ln) {
                 int col = nt * 16 + (ln & 15);
                 int mb = mt * 16 + ((ln >> 4) << 3);
#pragma unroll
                 for (int r = 0; r < 8; ++r) {
                   int m = mb + r;
                   float val = c[r];
                   if (col < Dm) {
                     tmp[(size_t)m * HS + col] = val;
                   } else if (col < 2 * Dm) {
                     int rc = col - Dm;
                     kcL[((size_t)(m * Hh + (rc >> 5)) * TMAX + t) * DH + (rc & 31)] = val;
                   } else {
                     int rc = col - 2 * Dm;
                     vcL[((size_t)(m * Hh + (rc >> 5)) * TMAX + t) * DH + (rc & 31)] = val;
                   }
                 }
               });
      __threadfence();
      __syncthreads();

      // causal attention, online softmax, TWO threads per (b, head):
      // thread p=0 scans [0, mid), p=1 scans [mid, t]; merged via LDS scratch.
      {
        const int bh = tid & 255;
        const int p  = tid >> 8;              // 0 or 1 (BLOCK == 512)
        const int b = bh >> 3, hh = bh & 7;
        float* qrow = tmp + (size_t)b * HS + hh * DH;
        float q[DH], o[DH];
#pragma unroll
        for (int d = 0; d < DH; ++d) { q[d] = qrow[d]; o[d] = 0.f; }
        float mx = -1e30f, sum = 0.f;
        const int mid = (t + 2) >> 1;         // ceil((t+1)/2)
        const int lo = p ? mid : 0;
        const int hi = p ? (t + 1) : mid;
        const float* kbase = kcL + (size_t)(b * Hh + hh) * TMAX * DH;
        const float* vbase = vcL + (size_t)(b * Hh + hh) * TMAX * DH;
        for (int tp2 = lo; tp2 < hi; ++tp2) {
          const float* kv = kbase + (size_t)tp2 * DH;
          float s = 0.f;
#pragma unroll
          for (int d = 0; d < DH; ++d) s += q[d] * kv[d];
          s *= 0.17677669529663687f;          // 1/sqrt(32)
          float nmx = fmaxf(mx, s);
          float corr = __expf(mx - nmx);
          float w = __expf(s - nmx);
          sum = sum * corr + w;
          const float* vv = vbase + (size_t)tp2 * DH;
#pragma unroll
          for (int d = 0; d < DH; ++d) o[d] = o[d] * corr + w * vv[d];
          mx = nmx;
        }
        if (p) {  // publish partial state
          float* sc = ascr + (size_t)bh * 36;
#pragma unroll
          for (int d = 0; d < DH; ++d) sc[d] = o[d];
          sc[32] = mx;
          sc[33] = sum;
        }
        __syncthreads();
        if (p == 0) {  // merge the two softmax partitions
          const float* sc = ascr + (size_t)bh * 36;
          float m2 = sc[32], s2 = sc[33];
          float nm = fmaxf(mx, m2);
          float c1 = __expf(mx - nm), c2 = __expf(m2 - nm);
          float inv = 1.f / (sum * c1 + s2 * c2);
#pragma unroll
          for (int d = 0; d < DH; ++d) qrow[d] = (o[d] * c1 + sc[d] * c2) * inv;
        }
      }
      __syncthreads();

      // h = ln1(h + attn @ Wo + bo)
      gemmA<8>(aT, A.wpack + OFF_DEC_WO + l * SZ_WO, 0, 16, A.dec_bo + l * Dm,
               [&](v8f c, int mt, int nt, int ln) {
                 int col = nt * 16 + (ln & 15);
                 int mb = mt * 16 + ((ln >> 4) << 3);
#pragma unroll
                 for (int r = 0; r < 8; ++r) h[(size_t)(mb + r) * HS + col] += c[r];
               });
      layernorm(h, A.dec_ln1_g + l * Dm, A.dec_ln1_b + l * Dm);

      // h = ln2(h + cross[l])   (residual add fused into LN pass)
      layernorm_add(h, A.dec_ln2_g + l * Dm, A.dec_ln2_b + l * Dm,
                    A.cross + (size_t)l * Bb * Dm);

      // FFN
      gemmA<8>(aH, A.wpack + OFF_DEC_W1 + l * SZ_W1, 0, 128, A.dec_b1 + l * Ff,
               [&](v8f c, int mt, int nt, int ln) {
                 int col = nt * 16 + (ln & 15);
                 int mb = mt * 16 + ((ln >> 4) << 3);
#pragma unroll
                 for (int r = 0; r < 8; ++r)
                   ffn[(size_t)(mb + r) * FS + col] = (_Float16)fmaxf(c[r], 0.f);
               });
      __syncthreads();
      gemmW2(ffn, A.wpack + OFF_DEC_W2 + l * SZ_W2, A.dec_b2 + l * Dm,
             [&](v8f c, int mt, int nt, int ln) {
               int col = nt * 16 + (ln & 15);
               int mb = mt * 16 + ((ln >> 4) << 3);
#pragma unroll
               for (int r = 0; r < 8; ++r) h[(size_t)(mb + r) * HS + col] += c[r];
             });
      layernorm(h, A.dec_ln3_g + l * Dm, A.dec_ln3_b + l * Dm);
    }

    // action = h @ W_out + b_out : wave per batch row (2 rows/wave), shuffle-reduced
    {
      const int wv = tid >> 5, lane = tid & 31;
#pragma unroll
      for (int rr = 0; rr < 2; ++rr) {
        const int w = wv + rr * 16;
        const float* row = h + (size_t)w * HS;
        float acc[7];
#pragma unroll
        for (int j = 0; j < 7; ++j) acc[j] = 0.f;
#pragma unroll
        for (int i = 0; i < 8; ++i) {
          int d = lane + i * 32;
          float x = row[d];
#pragma unroll
          for (int j = 0; j < 7; ++j) acc[j] += x * A.W_out[d * 7 + j];
        }
#pragma unroll
        for (int j = 0; j < 7; ++j) {
          float r = waveRed(acc[j]);
          if (lane == 0) {
            float s = r + A.b_out[j];
            A.out[((size_t)w * T + t) * 7 + j] = s;
            tmp[(size_t)w * HS + j] = s;
          }
        }
      }
    }
    __syncthreads();

    // next token = action @ W_act + b_act
    for (int i = tid; i < Bb * Dm; i += BLOCK) {
      int b = i >> 8, d = i & 255;
      float s = A.b_act[d];
#pragma unroll
      for (int j = 0; j < 7; ++j) s += tmp[(size_t)b * HS + j] * A.W_act[j * Dm + d];
      tok[(size_t)b * HS + d] = s;
    }
    __syncthreads();
  }
}

// ---------------------------------------------------------------------------
// Host launcher
// ---------------------------------------------------------------------------
extern "C" void kernel_launch(void* const* d_in, const int* in_sizes, int n_in,
                              void* d_out, int out_size, void* d_ws, size_t ws_size,
                              hipStream_t stream) {
  (void)in_sizes; (void)n_in; (void)out_size; (void)ws_size;

  char* ws = (char*)d_ws;
  _Float16* wpack = (_Float16*)ws;
  size_t off = (W_TOTAL * 2 + 255) & ~(size_t)255;
  float* pe    = (float*)(ws + off); off += (size_t)TMAX * Dm * 4;
  float* cross = (float*)(ws + off); off += (size_t)Lnum * Bb * Dm * 4;
  float* kc    = (float*)(ws + off); off += (size_t)Lnum * Bb * Hh * TMAX * DH * 4;
  float* vc    = (float*)(ws + off); off += (size_t)Lnum * Bb * Hh * TMAX * DH * 4;

  pe_kernel<<<200, 256, 0, stream>>>(pe);

  struct Job { const float* s; _Float16* d; int K, N; };
  const float* encWqkv = (const float*)d_in[11];
  const float* encWo   = (const float*)d_in[13];
  const float* encW1   = (const float*)d_in[15];
  const float* encW2   = (const float*)d_in[17];
  const float* decWqkv = (const float*)d_in[23];
  const float* decWo   = (const float*)d_in[25];
  const float* decCkv  = (const float*)d_in[27];
  const float* decCwo  = (const float*)d_in[29];
  const float* decW1   = (const float*)d_in[31];
  const float* decW2   = (const float*)d_in[33];

  Job jobs[20];
  int nj = 0;
  for (int l = 0; l < Lnum; ++l) {
    jobs[nj++] = {encWqkv + l * SZ_QKV, wpack + OFF_ENC_WQKV + l * SZ_QKV, Dm, 3 * Dm};
    jobs[nj++] = {encWo   + l * SZ_WO,  wpack + OFF_ENC_WO   + l * SZ_WO,  Dm, Dm};
    jobs[nj++] = {encW1   + l * SZ_W1,  wpack + OFF_ENC_W1   + l * SZ_W1,  Dm, Ff};
    jobs[nj++] = {encW2   + l * SZ_W2,  wpack + OFF_ENC_W2   + l * SZ_W2,  Ff, Dm};
    jobs[nj++] = {decWqkv + l * SZ_QKV, wpack + OFF_DEC_WQKV + l * SZ_QKV, Dm, 3 * Dm};
    jobs[nj++] = {decWo   + l * SZ_WO,  wpack + OFF_DEC_WO   + l * SZ_WO,  Dm, Dm};
    jobs[nj++] = {decCkv  + l * SZ_CKV, wpack + OFF_DEC_CWKV + l * SZ_CKV, Dm, 2 * Dm};
    jobs[nj++] = {decCwo  + l * SZ_WO,  wpack + OFF_DEC_CWO  + l * SZ_WO,  Dm, Dm};
    jobs[nj++] = {decW1   + l * SZ_W1,  wpack + OFF_DEC_W1   + l * SZ_W1,  Dm, Ff};
    jobs[nj++] = {decW2   + l * SZ_W2,  wpack + OFF_DEC_W2   + l * SZ_W2,  Ff, Dm};
  }
  for (int j = 0; j < nj; ++j)
    pack_b<<<256, 256, 0, stream>>>(jobs[j].s, jobs[j].d, jobs[j].K, jobs[j].N);

  NetArgs A;
  A.ja = (const float*)d_in[0];
  A.jp = (const float*)d_in[1];
  A.tp = (const float*)d_in[2];
  A.maxlen = (const int*)d_in[3];
  A.W_in  = (const float*)d_in[4];
  A.b_in  = (const float*)d_in[5];
  A.W_act = (const float*)d_in[6];
  A.b_act = (const float*)d_in[7];
  A.W_out = (const float*)d_in[8];
  A.b_out = (const float*)d_in[9];
  A.start_token = (const float*)d_in[10];
  A.enc_bqkv  = (const float*)d_in[12];
  A.enc_bo    = (const float*)d_in[14];
  A.enc_b1    = (const float*)d_in[16];
  A.enc_b2    = (const float*)d_in[18];
  A.enc_ln1_g = (const float*)d_in[19];
  A.enc_ln1_b = (const float*)d_in[20];
  A.enc_ln2_g = (const float*)d_in[21];
  A.enc_ln2_b = (const float*)d_in[22];
  A.dec_bqkv  = (const float*)d_in[24];
  A.dec_bo    = (const float*)d_in[26];
  A.dec_cbkv  = (const float*)d_in[28];
  A.dec_cbo   = (const float*)d_in[30];
  A.dec_b1    = (const float*)d_in[32];
  A.dec_b2    = (const float*)d_in[34];
  A.dec_ln1_g = (const float*)d_in[35];
  A.dec_ln1_b = (const float*)d_in[36];
  A.dec_ln2_g = (const float*)d_in[37];
  A.dec_ln2_b = (const float*)d_in[38];
  A.dec_ln3_g = (const float*)d_in[39];
  A.dec_ln3_b = (const float*)d_in[40];
  A.wpack = wpack;
  A.pe = pe;
  A.cross = cross;
  A.kc = kc;
  A.vc = vc;
  A.out = (float*)d_out;

  (void)hipFuncSetAttribute((const void*)&traj_main,
                            hipFuncAttributeMaxDynamicSharedMemorySize,
                            (int)LDS_BYTES);
  traj_main<<<1, BLOCK, LDS_BYTES, stream>>>(A);
}
